// HOPNet_Layer_62483184222900
// MI455X (gfx1250) — matrix operand order repaired and checked by reading, not compiled, not run on hardware
//
#include <hip/hip_runtime.h>
#include <hip/hip_bf16.h>
#include <cstdint>

// ---------------------------------------------------------------------------
// Problem constants (match reference)
// ---------------------------------------------------------------------------
static constexpr int kC   = 128;
static constexpr int kN0  = 50000;
static constexpr int kN1  = 150000;
static constexpr int kN2  = 100000;
static constexpr int kN3  = 75000;
static constexpr int kN4  = 512;
static constexpr int kNNZ02 = 300000;
static constexpr int kNNZ12 = 300000;
static constexpr int kNNZ23 = 2 * kN3;

typedef __attribute__((ext_vector_type(16))) __bf16 v16bf;
typedef __attribute__((ext_vector_type(2)))  __bf16 v2bf;
typedef __attribute__((ext_vector_type(2)))  float  v2f;
typedef __attribute__((ext_vector_type(8)))  float  v8f;
typedef __attribute__((__vector_size__(4 * sizeof(int)))) int vi4;

// Optional CDNA5 async global->LDS path (ASYNCcnt), guarded so the file
// compiles on toolchains that lack the builtins.
#if defined(__has_builtin)
#if __has_builtin(__builtin_amdgcn_global_load_async_to_lds_b128) && \
    __has_builtin(__builtin_amdgcn_s_wait_asynccnt)
#define USE_ASYNC_LDS 1
#endif
#endif
#ifndef USE_ASYNC_LDS
#define USE_ASYNC_LDS 0
#endif

// f32 -> bf16 (single); native cast lowers to v_cvt_pk_bf16_f32 on gfx1250
__device__ __forceinline__ __bf16 f2bf(float f) {
    return (__bf16)f;
}
// pack two f32 -> one dword {hi:bf16(b), lo:bf16(a)}; vector fptrunc lowers
// to a single v_cvt_pk_bf16_f32 with both sources
__device__ __forceinline__ unsigned bfpair(float a, float b) {
    v2f f; f[0] = a; f[1] = b;
    union { v2bf v; unsigned u; } z;
    z.v = __builtin_convertvector(f, v2bf);
    return z.u;
}

struct Seg  { const float* p; int ld; int off; };
struct Segs4 { Seg s[4]; };

// ---------------------------------------------------------------------------
// Weight pre-pack: W [din][128] f32 row-major  ->  WT [128][din/2] dwords,
// WT[c][kq] = {bf16(W[2kq+1][c]) , bf16(W[2kq][c])}  (transposed + packed)
// ---------------------------------------------------------------------------
__global__ void pack_weights_kernel(const float* __restrict__ W,
                                    unsigned* __restrict__ WT, int din)
{
    long n = (long)kC * (din / 2);
    long g = (long)blockIdx.x * blockDim.x + threadIdx.x;
    if (g >= n) return;
    int c  = (int)(g / (din / 2));
    int kq = (int)(g % (din / 2));
    WT[g] = bfpair(W[(long)(2 * kq) * kC + c], W[(long)(2 * kq + 1) * kC + c]);
}

// ---------------------------------------------------------------------------
// Fused 2-layer MLP: out = relu(X@W1 + b1) @ W2 + b2
// X = column-concat of NSEG segments of width 128 (fp32, packed to bf16).
// Block: 256 threads (8 waves) -> 128-row x 128-col output tile.
// Wave tile: 32 rows x 64 cols (2x4 accumulators, B fragments reused x2).
// ---------------------------------------------------------------------------
template <int NSEG>
__global__ __launch_bounds__(256) void mlp_kernel(
    Segs4 segs,
    const unsigned* __restrict__ W1T, const float* __restrict__ b1,
    const unsigned* __restrict__ W2T, const float* __restrict__ b2,
    float* __restrict__ out, int N)
{
    constexpr int DIN = 128 * NSEG;
    __shared__ unsigned Xs_u[128][16];        // 128 rows x 32 bf16 (activations)
    __shared__ unsigned Ws_u[128][16];        // 128 cols x 32 bf16 (weightsT)
    __shared__ unsigned Hs_u[128][kC / 2 + 4];// hidden, bf16, padded rows

    const int tid  = threadIdx.x;
    const int wave = tid >> 5;
    const int lane = tid & 31;
    const int half = lane >> 4;
    const int l15  = lane & 15;
    const int rbase = (wave >> 1) * 32;       // wave row-group
    const int cbase = (wave & 1) * 64;        // wave col-group
    const long rowBase = (long)blockIdx.x * 128;

    v8f acc[2][4];
#pragma unroll
    for (int a = 0; a < 2; ++a)
#pragma unroll
        for (int b = 0; b < 4; ++b)
#pragma unroll
            for (int j = 0; j < 8; ++j) acc[a][b][j] = 0.f;

    // ---------------- stage 1: hidden = relu(X @ W1 + b1) ----------------
    for (int k0 = 0; k0 < DIN; k0 += 32) {
        const int segIdx = k0 >> 7;
        const int kin    = k0 & 127;
        const float* sp  = segs.s[segIdx].p;
        const int ld     = segs.s[segIdx].ld;
        const int off    = segs.s[segIdx].off;

        // activations: 128x32 f32, each thread: 4x float4 load + uint2 LDS store
#pragma unroll
        for (int i = 0; i < 4; ++i) {
            int idx = tid + i * 256;          // 0..1023 float4 pieces
            int r   = idx >> 3;
            int q   = idx & 7;                // float4 within row
            long gr = rowBase + r;
            float4 v = {0.f, 0.f, 0.f, 0.f};
            if (gr < N) v = *(const float4*)(sp + gr * (long)ld + off + kin + q * 4);
            uint2 t; t.x = bfpair(v.x, v.y); t.y = bfpair(v.z, v.w);
            *(uint2*)&Xs_u[r][q * 2] = t;
        }
        // weights: pre-packed bf16, 128 rows x 64B -> copy 16B pieces
#pragma unroll
        for (int i = 0; i < 2; ++i) {
            int idx = tid + i * 256;          // 0..511 uint4 pieces
            int c   = idx >> 2;
            int pc  = idx & 3;
            const uint4* src = (const uint4*)(W1T + (long)c * (DIN / 2) + (k0 >> 1)) + pc;
#if USE_ASYNC_LDS
            __builtin_amdgcn_global_load_async_to_lds_b128(
                (__attribute__((address_space(1))) vi4*)src,
                (__attribute__((address_space(3))) vi4*)&Ws_u[c][pc * 4], 0, 0);
#else
            *(uint4*)&Ws_u[c][pc * 4] = *src;
#endif
        }
#if USE_ASYNC_LDS
        __builtin_amdgcn_s_wait_asynccnt(0);
#endif
        __syncthreads();

        v16bf a[2];
#pragma unroll
        for (int rt = 0; rt < 2; ++rt) {
            const __bf16* xr = (const __bf16*)&Xs_u[rbase + rt * 16 + l15][0];
#pragma unroll
            for (int j = 0; j < 8; ++j) a[rt][j]     = xr[half * 8 + j];
#pragma unroll
            for (int j = 0; j < 8; ++j) a[rt][8 + j] = xr[16 + half * 8 + j];
        }
#pragma unroll
        for (int ct = 0; ct < 4; ++ct) {
            v16bf b;
            const __bf16* br = (const __bf16*)&Ws_u[cbase + ct * 16 + l15][0];
#pragma unroll
            for (int j = 0; j < 16; ++j) b[j] = br[(j < 8 ? 0 : 8) + half * 8 + j];
            acc[0][ct] = __builtin_amdgcn_wmma_f32_16x16x32_bf16(
                false, a[0], false, b, (short)0, acc[0][ct], false, false);
            acc[1][ct] = __builtin_amdgcn_wmma_f32_16x16x32_bf16(
                false, a[1], false, b, (short)0, acc[1][ct], false, false);
        }
        __syncthreads();
    }

    // bias + relu -> Hs (bf16); reset accumulators
#pragma unroll
    for (int rt = 0; rt < 2; ++rt)
#pragma unroll
        for (int ct = 0; ct < 4; ++ct) {
            const int col = cbase + ct * 16 + l15;
            const float bb = b1[col];
#pragma unroll
            for (int r = 0; r < 8; ++r) {
                const int row = rbase + rt * 16 + half * 8 + r;
                float v = acc[rt][ct][r] + bb;
                ((__bf16*)&Hs_u[row][0])[col] = f2bf(v > 0.f ? v : 0.f);
                acc[rt][ct][r] = 0.f;
            }
        }
    __syncthreads();

    // ---------------- stage 2: out = hidden @ W2 + b2 ----------------
    for (int k0 = 0; k0 < kC; k0 += 32) {
#pragma unroll
        for (int i = 0; i < 2; ++i) {
            int idx = tid + i * 256;
            int c   = idx >> 2;
            int pc  = idx & 3;
            const uint4* src = (const uint4*)(W2T + (long)c * 64 + (k0 >> 1)) + pc;
#if USE_ASYNC_LDS
            __builtin_amdgcn_global_load_async_to_lds_b128(
                (__attribute__((address_space(1))) vi4*)src,
                (__attribute__((address_space(3))) vi4*)&Ws_u[c][pc * 4], 0, 0);
#else
            *(uint4*)&Ws_u[c][pc * 4] = *src;
#endif
        }
#if USE_ASYNC_LDS
        __builtin_amdgcn_s_wait_asynccnt(0);
#endif
        __syncthreads();

        v16bf a[2];
#pragma unroll
        for (int rt = 0; rt < 2; ++rt) {
            const __bf16* xr = (const __bf16*)&Hs_u[rbase + rt * 16 + l15][0];
#pragma unroll
            for (int j = 0; j < 8; ++j) a[rt][j]     = xr[k0 + half * 8 + j];
#pragma unroll
            for (int j = 0; j < 8; ++j) a[rt][8 + j] = xr[k0 + 16 + half * 8 + j];
        }
#pragma unroll
        for (int ct = 0; ct < 4; ++ct) {
            v16bf b;
            const __bf16* br = (const __bf16*)&Ws_u[cbase + ct * 16 + l15][0];
#pragma unroll
            for (int j = 0; j < 16; ++j) b[j] = br[(j < 8 ? 0 : 8) + half * 8 + j];
            acc[0][ct] = __builtin_amdgcn_wmma_f32_16x16x32_bf16(
                false, a[0], false, b, (short)0, acc[0][ct], false, false);
            acc[1][ct] = __builtin_amdgcn_wmma_f32_16x16x32_bf16(
                false, a[1], false, b, (short)0, acc[1][ct], false, false);
        }
        __syncthreads();
    }

#pragma unroll
    for (int rt = 0; rt < 2; ++rt)
#pragma unroll
        for (int ct = 0; ct < 4; ++ct) {
            const int col = cbase + ct * 16 + l15;
            const float bb = b2[col];
#pragma unroll
            for (int r = 0; r < 8; ++r) {
                const long gr = rowBase + rbase + rt * 16 + half * 8 + r;
                if (gr < N) out[gr * kC + col] = acc[rt][ct][r] + bb;
            }
        }
}

// ---------------------------------------------------------------------------
// Sparse message-passing helper kernels
// ---------------------------------------------------------------------------
__global__ void zero_kernel(float* p, long n) {
    long i = (long)blockIdx.x * blockDim.x + threadIdx.x;
    long stride = (long)gridDim.x * blockDim.x;
    for (; i < n; i += stride) p[i] = 0.f;
}

__global__ void scatter_add_kernel(const float* __restrict__ msgs,
                                   const int* __restrict__ rows,
                                   const int* __restrict__ cols,
                                   const float* __restrict__ vals,
                                   float* __restrict__ out, int nnz)
{
    long g = (long)blockIdx.x * blockDim.x + threadIdx.x;
    if (g >= (long)nnz * kC) return;
    int e = (int)(g >> 7), c = (int)(g & 127);
    int r = rows[e], co = cols[e];
    atomicAdd(&out[(long)co * kC + c], vals[e] * msgs[(long)r * kC + c]);
}

__global__ void count_kernel(const int* __restrict__ cols, float* __restrict__ cnt, int nnz) {
    int e = blockIdx.x * blockDim.x + threadIdx.x;
    if (e < nnz) atomicAdd(&cnt[cols[e]], 1.f);
}

__global__ void meandiv_kernel(float* __restrict__ buf, const float* __restrict__ cnt, int rowsN) {
    long g = (long)blockIdx.x * blockDim.x + threadIdx.x;
    if (g >= (long)rowsN * kC) return;
    buf[g] /= fmaxf(cnt[g >> 7], 1.f);
}

__global__ void sign_count_kernel(const float* __restrict__ vals,
                                  const int* __restrict__ faces,
                                  float* __restrict__ hp, float* __restrict__ hn, int nnz)
{
    int e = blockIdx.x * blockDim.x + threadIdx.x;
    if (e >= nnz) return;
    float v = vals[e]; int f = faces[e];
    if (v > 0.f) atomicAdd(&hp[f], 1.f);
    if (v < 0.f) atomicAdd(&hn[f], 1.f);
}

__global__ void pm_scatter_kernel(const float* __restrict__ f23,
                                  const float* __restrict__ hp,
                                  const float* __restrict__ hn,
                                  const int* __restrict__ faces,
                                  const int* __restrict__ colls,
                                  float* __restrict__ plusb,
                                  float* __restrict__ minusb, int nnz)
{
    long g = (long)blockIdx.x * blockDim.x + threadIdx.x;
    if (g >= (long)nnz * kC) return;
    int e = (int)(g >> 7), c = (int)(g & 127);
    int f = faces[e], co = colls[e];
    float m  = f23[(long)f * kC + c];
    float mp = (hp[f] > 0.f) ? m : 0.f;
    float mn = (hn[f] > 0.f) ? m : 0.f;
    atomicAdd(&plusb [(long)co * (2 * kC) + c],      mn);
    atomicAdd(&plusb [(long)co * (2 * kC) + kC + c], mp);
    atomicAdd(&minusb[(long)co * (2 * kC) + c],      mp);
    atomicAdd(&minusb[(long)co * (2 * kC) + kC + c], mn);
}

__global__ void back_scatter_kernel(const float* __restrict__ h3pp,
                                    const float* __restrict__ h3pm,
                                    const float* __restrict__ vals,
                                    const int* __restrict__ faces,
                                    const int* __restrict__ colls,
                                    float* __restrict__ m32, int nnz)
{
    long g = (long)blockIdx.x * blockDim.x + threadIdx.x;
    if (g >= (long)nnz * kC) return;
    int e = (int)(g >> 7), c = (int)(g & 127);
    int f = faces[e], co = colls[e];
    float v = vals[e];
    float gsum = 0.f;
    if (v > 0.f) gsum += h3pp[(long)co * kC + c];
    if (v < 0.f) gsum += h3pm[(long)co * kC + c];
    atomicAdd(&m32[(long)f * kC + c], gsum);
}

// ---------------------------------------------------------------------------
// Driver
// ---------------------------------------------------------------------------
extern "C" void kernel_launch(void* const* d_in, const int* in_sizes, int n_in,
                              void* d_out, int out_size, void* d_ws, size_t ws_size,
                              hipStream_t stream)
{
    (void)in_sizes; (void)n_in; (void)out_size; (void)ws_size;

    const float* h0  = (const float*)d_in[0];
    const float* h1  = (const float*)d_in[1];
    const float* h2  = (const float*)d_in[2];
    const float* h3m = (const float*)d_in[3];
    const float* h3p = (const float*)d_in[4];
    const float* h4  = (const float*)d_in[5];
    const int*   b02r = (const int*)d_in[6];
    const int*   b02c = (const int*)d_in[7];
    const float* b02v = (const float*)d_in[8];
    const int*   b04r = (const int*)d_in[9];
    const int*   b04c = (const int*)d_in[10];
    const float* b04v = (const float*)d_in[11];
    const int*   b12r = (const int*)d_in[12];
    const int*   b12c = (const int*)d_in[13];
    const float* b12v = (const float*)d_in[14];
    const int*   b23f = (const int*)d_in[15];
    const int*   b23c = (const int*)d_in[16];
    const float* b23v = (const float*)d_in[17];
    const int*   b24r = (const int*)d_in[18];
    const int*   b24c = (const int*)d_in[19];
    const float* b24v = (const float*)d_in[20];
    auto PW = [&](int i, int j) { return (const float*)d_in[21 + 4 * i + j]; };

    float* out    = (float*)d_out;
    float* o_h0p  = out;
    float* o_h1   = o_h0p  + (long)kN0 * kC;
    float* o_h2pp = o_h1   + (long)kN1 * kC;
    float* o_h3m  = o_h2pp + (long)kN2 * kC;
    float* o_h3p  = o_h3m  + (long)kN3 * kC;
    float* o_h4pp = o_h3p  + (long)kN3 * kC;

    float* ws = (float*)d_ws;
    long off = 0;
    float* W_T   = ws + off; off += (long)kN1 * kC;      // msg buf / PLUS / T24 / T04
    float* W_M0  = ws + off; off += (long)kN2 * kC;      // m0to2 / m3to2
    float* W_M1  = ws + off; off += (long)kN2 * kC;      // m1to2 / m2to3_f
    float* W_M4  = ws + off; off += (long)kN2 * kC;      // m4to2
    float* W_H2P = ws + off; off += (long)kN2 * kC;      // h2p
    float* W_MIN = ws + off; off += (long)kN3 * 2 * kC;  // minus buffer
    float* W_M40 = ws + off; off += (long)kN0 * kC;      // m4to0
    float* W_HP  = ws + off; off += kN2;
    float* W_HN  = ws + off; off += kN2;
    float* W_M24 = ws + off; off += (long)kN4 * kC;
    float* W_C24 = ws + off; off += kN4;
    float* W_H4P = ws + off; off += (long)kN4 * kC;
    float* W_M04 = ws + off; off += (long)kN4 * kC;
    float* W_C04 = ws + off; off += kN4;
    float* W_T40 = ws + off; off += (long)kN4 * kC;
    float* W_PLUS = W_T;     // N3*2C == N1*C, lifetimes disjoint

    // bf16 packed-transposed weight arena (16B aligned)
    off = (off + 3) & ~3L;
    static const int dins[13] = {128, 128, 128, 512, 128, 384, 256, 128, 256, 128, 128, 256, 256};
    unsigned* wArena = (unsigned*)(ws + off);
    unsigned* W1T[13]; unsigned* W2T[13];
    long woff = 0;
    for (int i = 0; i < 13; ++i) {
        W1T[i] = wArena + woff; woff += (long)kC * (dins[i] / 2);
        W2T[i] = wArena + woff; woff += (long)kC * 64;
    }
    for (int i = 0; i < 13; ++i) {
        long n1 = (long)kC * (dins[i] / 2);
        pack_weights_kernel<<<(unsigned)((n1 + 255) / 256), 256, 0, stream>>>(PW(i, 0), W1T[i], dins[i]);
        long n2 = (long)kC * 64;
        pack_weights_kernel<<<(unsigned)((n2 + 255) / 256), 256, 0, stream>>>(PW(i, 2), W2T[i], kC);
    }

    auto zero = [&](float* p, long n) {
        zero_kernel<<<1024, 256, 0, stream>>>(p, n);
    };
    auto scat = [&](const float* msgs, const int* rows, const int* cols,
                    const float* vals, float* dst, int nnz) {
        long tot = (long)nnz * kC;
        scatter_add_kernel<<<(unsigned)((tot + 255) / 256), 256, 0, stream>>>(
            msgs, rows, cols, vals, dst, nnz);
    };
    auto mlp = [&](int nseg, Segs4 sg, int pi, float* dst, int N) {
        dim3 g((N + 127) / 128), b(256);
        switch (nseg) {
        case 1: mlp_kernel<1><<<g, b, 0, stream>>>(sg, W1T[pi], PW(pi, 1), W2T[pi], PW(pi, 3), dst, N); break;
        case 2: mlp_kernel<2><<<g, b, 0, stream>>>(sg, W1T[pi], PW(pi, 1), W2T[pi], PW(pi, 3), dst, N); break;
        case 3: mlp_kernel<3><<<g, b, 0, stream>>>(sg, W1T[pi], PW(pi, 1), W2T[pi], PW(pi, 3), dst, N); break;
        default: mlp_kernel<4><<<g, b, 0, stream>>>(sg, W1T[pi], PW(pi, 1), W2T[pi], PW(pi, 3), dst, N); break;
        }
    };
    auto S1 = [&](const float* a) { Segs4 s{}; s.s[0] = {a, kC, 0}; s.s[1] = s.s[2] = s.s[3] = s.s[0]; return s; };
    auto S2 = [&](const float* a, const float* b) {
        Segs4 s{}; s.s[0] = {a, kC, 0}; s.s[1] = {b, kC, 0}; s.s[2] = s.s[3] = s.s[1]; return s; };
    auto S4 = [&](const float* a, const float* b, const float* c, const float* d) {
        Segs4 s{}; s.s[0] = {a, kC, 0}; s.s[1] = {b, kC, 0}; s.s[2] = {c, kC, 0}; s.s[3] = {d, kC, 0}; return s; };

    // ---- level-2 messages from vertices / edges / objects ----
    mlp(1, S1(h0), 0, W_T, kN0);                       // 0to2
    zero(W_M0, (long)kN2 * kC);
    scat(W_T, b02r, b02c, b02v, W_M0, kNNZ02);

    mlp(1, S1(h1), 1, W_T, kN1);                       // 1to2
    zero(W_M1, (long)kN2 * kC);
    scat(W_T, b12r, b12c, b12v, W_M1, kNNZ12);

    mlp(1, S1(h4), 2, W_T, kN4);                       // 4to2
    zero(W_M4, (long)kN2 * kC);
    scat(W_T, b24c, b24r, b24v, W_M4, kN2);            // b24.T

    mlp(4, S4(h2, W_M0, W_M1, W_M4), 3, W_H2P, kN2);   // p2 -> h2p

    // ---- sign-split face -> collision messages ----
    mlp(1, S1(W_H2P), 4, W_M1, kN2);                   // 2to3 -> m2to3_f (reuse M1)
    zero(W_HP, kN2); zero(W_HN, kN2);
    sign_count_kernel<<<(kNNZ23 + 255) / 256, 256, 0, stream>>>(b23v, b23f, W_HP, W_HN, kNNZ23);
    zero(W_PLUS, (long)kN3 * 2 * kC);
    zero(W_MIN,  (long)kN3 * 2 * kC);
    {
        long tot = (long)kNNZ23 * kC;
        pm_scatter_kernel<<<(unsigned)((tot + 255) / 256), 256, 0, stream>>>(
            W_M1, W_HP, W_HN, b23f, b23c, W_PLUS, W_MIN, kNNZ23);
    }
    {
        Segs4 sp{}; sp.s[0] = {h3p, kC, 0}; sp.s[1] = {W_PLUS, 2 * kC, 0};
        sp.s[2] = {W_PLUS, 2 * kC, kC}; sp.s[3] = sp.s[2];
        mlp(3, sp, 5, o_h3p, kN3);                     // p3 (plus)
        Segs4 sm{}; sm.s[0] = {h3m, kC, 0}; sm.s[1] = {W_MIN, 2 * kC, 0};
        sm.s[2] = {W_MIN, 2 * kC, kC}; sm.s[3] = sm.s[2];
        mlp(3, sm, 5, o_h3m, kN3);                     // p3 (minus)
    }

    // ---- collision -> face sign-masked scatter ----
    zero(W_M0, (long)kN2 * kC);                        // reuse as m3to2
    {
        long tot = (long)kNNZ23 * kC;
        back_scatter_kernel<<<(unsigned)((tot + 255) / 256), 256, 0, stream>>>(
            o_h3p, o_h3m, b23v, b23f, b23c, W_M0, kNNZ23);
    }
    mlp(2, S2(W_H2P, W_M0), 6, o_h2pp, kN2);           // 2p -> h2pp

    // ---- face -> object (mean) ----
    mlp(1, S1(o_h2pp), 7, W_T, kN2);                   // 2to4
    zero(W_M24, (long)kN4 * kC); zero(W_C24, kN4);
    scat(W_T, b24r, b24c, b24v, W_M24, kN2);
    count_kernel<<<(kN2 + 255) / 256, 256, 0, stream>>>(b24c, W_C24, kN2);
    meandiv_kernel<<<(unsigned)(((long)kN4 * kC + 255) / 256), 256, 0, stream>>>(W_M24, W_C24, kN4);
    mlp(2, S2(h4, W_M24), 8, W_H4P, kN4);              // p4 -> h4p

    // ---- vertex <-> object exchange ----
    mlp(1, S1(h0), 9, W_T, kN0);                       // 0to4
    zero(W_M04, (long)kN4 * kC); zero(W_C04, kN4);
    scat(W_T, b04r, b04c, b04v, W_M04, kN0);
    count_kernel<<<(kN0 + 255) / 256, 256, 0, stream>>>(b04c, W_C04, kN0);
    meandiv_kernel<<<(unsigned)(((long)kN4 * kC + 255) / 256), 256, 0, stream>>>(W_M04, W_C04, kN4);

    mlp(1, S1(W_H4P), 10, W_T40, kN4);                 // 4to0
    zero(W_M40, (long)kN0 * kC);
    scat(W_T40, b04c, b04r, b04v, W_M40, kN0);         // b04.T

    mlp(2, S2(h0, W_M40), 11, o_h0p, kN0);             // p0 -> h0p
    mlp(2, S2(W_H4P, W_M04), 12, o_h4pp, kN4);         // 4p -> h4pp

    // h1 passes through unchanged
    (void)hipMemcpyAsync(o_h1, h1, (size_t)kN1 * kC * sizeof(float),
                         hipMemcpyDeviceToDevice, stream);
}
